// MultiTokenAttention_34849364640472
// MI455X (gfx1250) — matrix operand order repaired and checked by reading, not compiled, not run on hardware
//
#include <hip/hip_runtime.h>

typedef __attribute__((ext_vector_type(16))) _Float16 v16h;
typedef __attribute__((ext_vector_type(8)))  _Float16 v8h;
typedef __attribute__((ext_vector_type(8)))  float    v8f;

#define B_  2
#define N_  2048
#define E_  1024
#define H_  16
#define D_  64

// ---------------------------------------------------------------------------
// Fragment-major LDS layouts so each WMMA operand loads as 2x ds_load_b128.
//
// A (row-major, lda multiple of 8 halfs, 16B-aligned rows):
//   lane L: row = row0 + (L&15); halfs [k0+hi .. +15],
//   hi = (L&16)?16:0  -> two contiguous 16B runs.
// B (fragment-major): element (k,n) stored at
//   (((n>>4)*Kg + (k>>4))*16 + (n&15))*24 + (k&15)
//   -> per lane a contiguous 16-half run (32B, 16B aligned); 48B column
//   stride = 12-bank lane stride -> conflict-free b128 reads.
// C/D f32: elem j -> row j (lanes<16) / j+8 (lanes>=16), col = lane&15.
// ---------------------------------------------------------------------------
__device__ __forceinline__ v16h lds_load_af(const _Float16* As, int lda,
                                            int row0, int k0, int lane) {
  const _Float16* p = As + (row0 + (lane & 15)) * lda + k0 + ((lane & 16) ? 16 : 0);
  union { v16h v; v8h h[2]; } u;
  u.h[0] = *(const v8h*)p;
  u.h[1] = *(const v8h*)(p + 8);
  return u.v;
}

__device__ __forceinline__ v16h lds_load_bf(const _Float16* Bs, int kgroups,
                                            int k0, int nb, int lane) {
  const int kh = (k0 >> 4) + ((lane & 16) ? 1 : 0);
  const _Float16* p = Bs + ((nb * kgroups + kh) * 16 + (lane & 15)) * 24;
  union { v16h v; v8h h[2]; } u;
  u.h[0] = *(const v8h*)p;
  u.h[1] = *(const v8h*)(p + 8);
  return u.v;
}

__device__ __forceinline__ int bfrag_addr(int k, int n, int kgroups) {
  return (((n >> 4) * kgroups + (k >> 4)) * 16 + (n & 15)) * 24 + (k & 15);
}

__device__ __forceinline__ v8h cvt8(const float* __restrict__ g) {
  v8h r;
#pragma unroll
  for (int i = 0; i < 8; ++i) r[i] = (_Float16)g[i];
  return r;
}

// ---------------------------------------------------------------------------
// Kernel 1: Y[M x Nout] = X[M x K] @ W[K x Nout] + bias   (f32 io, f16 WMMA)
// block = 128 threads (4 waves); block tile 64x64; wave tile 16x64.
// ---------------------------------------------------------------------------
__global__ __launch_bounds__(128) void gemm_xw_bias(
    const float* __restrict__ X, const float* __restrict__ W,
    const float* __restrict__ bias, float* __restrict__ Y,
    int M, int K, int Nout) {
  __shared__ __align__(16) _Float16 As[64 * 40];      // 64 rows x 32 k
  __shared__ __align__(16) _Float16 Bs[4 * 2 * 16 * 24];  // frag-major, Kg=2
  const int tid  = threadIdx.x;
  const int wave = tid >> 5;
  const int lane = tid & 31;
  const int row0 = blockIdx.y * 64;
  const int col0 = blockIdx.x * 64;

  v8f acc[4] = {};
  for (int kt = 0; kt < K; kt += 32) {
    // A: vectorized stage, thread converts 8 consecutive k of one row
    for (int idx = tid; idx < 64 * 4; idx += 128) {
      int r = idx >> 2, co = (idx & 3) * 8;
      *(v8h*)&As[r * 40 + co] = cvt8(&X[(size_t)(row0 + r) * K + kt + co]);
    }
    // B (W is k-major in global, n contiguous): scalar stage into frag layout
    for (int idx = tid; idx < 32 * 64; idx += 128) {
      int k = idx >> 6, n = idx & 63;
      Bs[bfrag_addr(k, n, 2)] = (_Float16)W[(size_t)(kt + k) * Nout + col0 + n];
    }
    __syncthreads();
    v16h a = lds_load_af(As, 40, wave * 16, 0, lane);
#pragma unroll
    for (int nb = 0; nb < 4; ++nb) {
      v16h bf = lds_load_bf(Bs, 2, 0, nb, lane);
      acc[nb] = __builtin_amdgcn_wmma_f32_16x16x32_f16(
          false, a, false, bf, (short)0, acc[nb], false, false);
    }
    __syncthreads();
  }
  const int coll = lane & 15;
  const int rhi  = (lane & 16) ? 8 : 0;
#pragma unroll
  for (int nb = 0; nb < 4; ++nb) {
    int col = col0 + nb * 16 + coll;
    float bv = bias[col];
#pragma unroll
    for (int j = 0; j < 8; ++j) {
      int row = row0 + wave * 16 + rhi + j;
      Y[(size_t)row * Nout + col] = acc[nb][j] + bv;
    }
  }
}

// ---------------------------------------------------------------------------
// Kernel 2: per (b,h) causal scores  S = (Q_h @ K_h^T) / 8, zero above diag.
// Both operands have d contiguous in global -> fully vectorized staging.
// ---------------------------------------------------------------------------
__global__ __launch_bounds__(128) void scores_kernel(
    const float* __restrict__ Qp, const float* __restrict__ Kp,
    float* __restrict__ S) {
  const int ktile = blockIdx.x, qtile = blockIdx.y, bh = blockIdx.z;
  const int b = bh >> 4, h = bh & 15;
  const int q0 = qtile * 64, k0 = ktile * 64;
  const int tid = threadIdx.x, wave = tid >> 5, lane = tid & 31;
  float* Stile = S + ((size_t)bh * N_ + q0) * N_ + k0;

  if (ktile > qtile) {  // fully above diagonal -> zeros (conv reads them)
    for (int idx = tid; idx < 64 * 64; idx += 128) {
      int r = idx >> 6, c = idx & 63;
      Stile[(size_t)r * N_ + c] = 0.0f;
    }
    return;
  }

  __shared__ __align__(16) _Float16 As[64 * 72];          // Q rows x d
  __shared__ __align__(16) _Float16 Bs[4 * 4 * 16 * 24];  // K^T frag-major, Kg=4
  for (int idx = tid; idx < 64 * 8; idx += 128) {
    int r = idx >> 3, co = (idx & 7) * 8;
    *(v8h*)&As[r * 72 + co] =
        cvt8(&Qp[(size_t)(b * N_ + q0 + r) * E_ + h * 64 + co]);
  }
  for (int idx = tid; idx < 64 * 8; idx += 128) {
    int kk = idx >> 3, dd = (idx & 7) * 8;  // element (k=dd.., n=kk)
    *(v8h*)&Bs[bfrag_addr(dd, kk, 4)] =
        cvt8(&Kp[(size_t)(b * N_ + k0 + kk) * E_ + h * 64 + dd]);
  }
  __syncthreads();

  v8f acc[4] = {};
#pragma unroll
  for (int ks = 0; ks < 64; ks += 32) {
    v16h a = lds_load_af(As, 72, wave * 16, ks, lane);
#pragma unroll
    for (int nb = 0; nb < 4; ++nb) {
      v16h bf = lds_load_bf(Bs, 4, ks, nb, lane);
      acc[nb] = __builtin_amdgcn_wmma_f32_16x16x32_f16(
          false, a, false, bf, (short)0, acc[nb], false, false);
    }
  }
  const int coll = lane & 15;
  const int rhi  = (lane & 16) ? 8 : 0;
#pragma unroll
  for (int nb = 0; nb < 4; ++nb) {
#pragma unroll
    for (int j = 0; j < 8; ++j) {
      int r = wave * 16 + rhi + j;
      int c = nb * 16 + coll;
      float v = ((k0 + c) <= (q0 + r)) ? acc[nb][j] * 0.125f : 0.0f;
      Stile[(size_t)r * N_ + c] = v;
    }
  }
}

// ---------------------------------------------------------------------------
// Kernel 3: grouped 6x11 conv over the (q,k) score plane + causal softmax.
// One workgroup = (b, h, 16 q-rows); 2ch x 21row x 138col f32 LDS tile per
// 128-col chunk; 18-wide register window -> 88 FMAs per 29 LDS reads.
// Fused row-max/exp/sum via __shfl_xor over the 16 threads of a row; writes
// e = exp(conv-m) in place and 1/sum to inv_s (folded into attn@V epilogue).
// ---------------------------------------------------------------------------
__global__ __launch_bounds__(256) void conv_softmax_kernel(
    const float* __restrict__ S, const float* __restrict__ Wc,
    float* __restrict__ Eb, float* __restrict__ inv_s) {
  const int q0 = blockIdx.x * 16;
  const int h  = blockIdx.y;
  const int b  = blockIdx.z;
  const int g  = h >> 1;
  const int tid = threadIdx.x;
  const int r  = tid >> 4;   // row within 16-row block
  const int cg = tid & 15;   // 8-column group
  const int q  = q0 + r;
  const int bh = b * H_ + h;
  const int qlast = q0 + 15;
  int kend = ((qlast >> 6) + 1) << 6;  // ceil64: zero-fill bound for attn@V
  if (kend > N_) kend = N_;

  __shared__ float Ls[2 * 21 * 138];
  __shared__ float Wl[2 * 6 * 11];
  for (int idx = tid; idx < 132; idx += 256) Wl[idx] = Wc[h * 132 + idx];

  const size_t chanBase = ((size_t)(b * H_ + g * 2)) * N_ * N_;
  float* Erow = Eb + ((size_t)bh * N_ + q) * N_;
  float mx = -3.0e38f;

  for (int kc = 0; kc < kend; kc += 128) {
    __syncthreads();
    for (int idx = tid; idx < 2 * 21 * 138; idx += 256) {
      int c   = idx / (21 * 138);
      int rem = idx - c * (21 * 138);
      int i   = rem / 138;
      int x   = rem - i * 138;
      int sr  = q0 - 5 + i;
      int sk  = kc - 5 + x;
      float v = 0.0f;
      if (sr >= 0 && sk >= 0 && sk < N_)
        v = S[chanBase + (size_t)c * N_ * N_ + (size_t)sr * N_ + sk];
      Ls[idx] = v;
    }
    __syncthreads();

    float acc[8];
#pragma unroll
    for (int j = 0; j < 8; ++j) acc[j] = 0.0f;
#pragma unroll
    for (int c = 0; c < 2; ++c) {
#pragma unroll
      for (int i = 0; i < 6; ++i) {
        const float* row = &Ls[((c * 21) + (r + i)) * 138 + cg * 8];
        float win[18];
#pragma unroll
        for (int t = 0; t < 18; ++t) win[t] = row[t];
        float wk[11];
#pragma unroll
        for (int jw = 0; jw < 11; ++jw) wk[jw] = Wl[(c * 6 + i) * 11 + jw];
#pragma unroll
        for (int j = 0; j < 8; ++j)
#pragma unroll
          for (int jw = 0; jw < 11; ++jw)
            acc[j] = fmaf(wk[jw], win[j + jw], acc[j]);
      }
    }
    const int kbase = kc + cg * 8;
#pragma unroll
    for (int j = 0; j < 8; ++j) {
      int k = kbase + j;
      if (k <= q) { Erow[k] = acc[j]; mx = fmaxf(mx, acc[j]); }
    }
  }

#pragma unroll
  for (int off = 8; off >= 1; off >>= 1)
    mx = fmaxf(mx, __shfl_xor(mx, off, 16));

  float s = 0.0f;
  for (int kc = 0; kc < kend; kc += 128) {
    const int kbase = kc + cg * 8;
#pragma unroll
    for (int j = 0; j < 8; ++j) {
      int k = kbase + j;
      if (k < kend) {
        if (k <= q) {
          float e = __expf(Erow[k] - mx);
          Erow[k] = e;
          s += e;
        } else {
          Erow[k] = 0.0f;  // zero so attn@V can read full 64-tiles
        }
      }
    }
  }
#pragma unroll
  for (int off = 8; off >= 1; off >>= 1)
    s += __shfl_xor(s, off, 16);
  if (cg == 0) inv_s[(size_t)bh * N_ + q] = 1.0f / s;
}

// ---------------------------------------------------------------------------
// Kernel 4: ctx_h = (E_bh @ V_h) * inv_s[row]; only k-tiles <= q-tile (causal).
// E tile stages vectorized (k contiguous in global); V is a true transpose ->
// scalar conflict-free stage into the fragment-major layout.
// ---------------------------------------------------------------------------
__global__ __launch_bounds__(128) void attnv_kernel(
    const float* __restrict__ Eb, const float* __restrict__ Vp,
    const float* __restrict__ inv_s, float* __restrict__ ctx) {
  const int qtile = blockIdx.x, bh = blockIdx.y;
  const int b = bh >> 4, h = bh & 15;
  const int q0 = qtile * 64;
  const int tid = threadIdx.x, wave = tid >> 5, lane = tid & 31;
  __shared__ __align__(16) _Float16 As[64 * 72];          // e rows x k
  __shared__ __align__(16) _Float16 Bs[4 * 4 * 16 * 24];  // V frag-major, Kg=4
  v8f acc[4] = {};

  for (int kt = 0; kt <= qtile; ++kt) {
    const int k0 = kt * 64;
    for (int idx = tid; idx < 64 * 8; idx += 128) {
      int r = idx >> 3, co = (idx & 7) * 8;
      *(v8h*)&As[r * 72 + co] =
          cvt8(&Eb[((size_t)bh * N_ + q0 + r) * N_ + k0 + co]);
    }
    for (int idx = tid; idx < 64 * 64; idx += 128) {
      int kk = idx >> 6, d = idx & 63;  // element (k=kk, n=d)
      Bs[bfrag_addr(kk, d, 4)] =
          (_Float16)Vp[(size_t)(b * N_ + k0 + kk) * E_ + h * 64 + d];
    }
    __syncthreads();
#pragma unroll
    for (int ks = 0; ks < 64; ks += 32) {
      v16h a = lds_load_af(As, 72, wave * 16, ks, lane);
#pragma unroll
      for (int nb = 0; nb < 4; ++nb) {
        v16h bf = lds_load_bf(Bs, 4, ks, nb, lane);
        acc[nb] = __builtin_amdgcn_wmma_f32_16x16x32_f16(
            false, a, false, bf, (short)0, acc[nb], false, false);
      }
    }
    __syncthreads();
  }
  const int coll = lane & 15;
  const int rhi  = (lane & 16) ? 8 : 0;
#pragma unroll
  for (int nb = 0; nb < 4; ++nb) {
#pragma unroll
    for (int j = 0; j < 8; ++j) {
      int qq = q0 + wave * 16 + rhi + j;
      int d  = nb * 16 + coll;
      ctx[(size_t)(b * N_ + qq) * E_ + h * 64 + d] =
          acc[nb][j] * inv_s[(size_t)bh * N_ + qq];
    }
  }
}

// ---------------------------------------------------------------------------
extern "C" void kernel_launch(void* const* d_in, const int* in_sizes, int n_in,
                              void* d_out, int out_size, void* d_ws, size_t ws_size,
                              hipStream_t stream) {
  (void)in_sizes; (void)n_in; (void)out_size; (void)ws_size;
  const float* query = (const float*)d_in[0];
  const float* key_  = (const float*)d_in[1];
  const float* value = (const float*)d_in[2];
  const float* Wq = (const float*)d_in[3];
  const float* bq = (const float*)d_in[4];
  const float* Wk = (const float*)d_in[5];
  const float* bk = (const float*)d_in[6];
  const float* Wv = (const float*)d_in[7];
  const float* bv = (const float*)d_in[8];
  const float* cw = (const float*)d_in[9];
  const float* Wo = (const float*)d_in[10];
  const float* bo = (const float*)d_in[11];

  char* ws = (char*)d_ws;
  size_t off = 0;
  auto carve = [&](size_t bytes) -> float* {
    float* p = (float*)(ws + off);
    off += (bytes + 255) & ~(size_t)255;
    return p;
  };
  const size_t projBytes  = (size_t)B_ * N_ * E_ * sizeof(float);       // 16 MB
  const size_t planeBytes = (size_t)B_ * H_ * N_ * N_ * sizeof(float);  // 512 MB
  float* qp   = carve(projBytes);
  float* kp   = carve(projBytes);
  float* vp   = carve(projBytes);
  float* ctx  = carve(projBytes);
  float* invs = carve((size_t)B_ * H_ * N_ * sizeof(float));
  float* S    = carve(planeBytes);
  float* Eb   = carve(planeBytes);

  const dim3 blk128(128);
  const dim3 ggemm(E_ / 64, (B_ * N_) / 64);

  gemm_xw_bias<<<ggemm, blk128, 0, stream>>>(query, Wq, bq, qp, B_ * N_, E_, E_);
  gemm_xw_bias<<<ggemm, blk128, 0, stream>>>(key_,  Wk, bk, kp, B_ * N_, E_, E_);
  gemm_xw_bias<<<ggemm, blk128, 0, stream>>>(value, Wv, bv, vp, B_ * N_, E_, E_);

  scores_kernel<<<dim3(N_ / 64, N_ / 64, B_ * H_), blk128, 0, stream>>>(qp, kp, S);

  conv_softmax_kernel<<<dim3(N_ / 16, H_, B_), dim3(256), 0, stream>>>(S, cw, Eb, invs);

  attnv_kernel<<<dim3(N_ / 64, B_ * H_), blk128, 0, stream>>>(Eb, vp, invs, ctx);

  gemm_xw_bias<<<ggemm, blk128, 0, stream>>>(ctx, Wo, bo, (float*)d_out, B_ * N_, E_, E_);
}
